// AttnBlock_6700148982236
// MI455X (gfx1250) — matrix operand order repaired and checked by reading, not compiled, not run on hardware
//
#include <hip/hip_runtime.h>
#include <hip/hip_bf16.h>

// ---------------------------------------------------------------------------
// AttnBlock for MI455X (gfx1250). All GEMM operands are kept K-contiguous so
// LDS staging is pure 16B copies -> async global->LDS (ASYNCcnt) when the
// builtin exists, with a spill-free synchronous fallback. Matrix math on
// v_wmma_f32_16x16x32_f16 (wave32).
// ---------------------------------------------------------------------------

typedef _Float16 h16_t;
typedef __attribute__((ext_vector_type(16))) _Float16 v16h;
typedef __attribute__((ext_vector_type(8)))  float    v8f;
typedef int v4i_ __attribute__((vector_size(16)));   // matches builtin param

union FragU { v16h v; uint4 u[2]; };
static_assert(sizeof(FragU) == 32, "frag size");

#define LDSTR 40   // padded LDS row stride in halves (80B, 16B aligned)

#define AS1 __attribute__((address_space(1)))
#define AS3 __attribute__((address_space(3)))

#if defined(__gfx1250__) && __has_builtin(__builtin_amdgcn_global_load_async_to_lds_b128)
#define HAVE_ASYNC 1
#else
#define HAVE_ASYNC 0
#endif

__device__ __forceinline__ void async_cp16(const h16_t* g, h16_t* l) {
#if HAVE_ASYNC
    __builtin_amdgcn_global_load_async_to_lds_b128(
        (AS1 v4i_*)(void*)g, (AS3 v4i_*)(void*)l, 0, 0);
#else
    *(uint4*)l = *(const uint4*)g;       // sync fallback: adjacent load+store
#endif
}

__device__ __forceinline__ void wait_async4() {   // allow next tile in flight
#if HAVE_ASYNC
#if __has_builtin(__builtin_amdgcn_s_wait_asynccnt)
    __builtin_amdgcn_s_wait_asynccnt(4);
#else
    asm volatile("s_wait_asynccnt 0x4" ::: "memory");
#endif
#endif
}
__device__ __forceinline__ void wait_async0() {
#if HAVE_ASYNC
#if __has_builtin(__builtin_amdgcn_s_wait_asynccnt)
    __builtin_amdgcn_s_wait_asynccnt(0);
#else
    asm volatile("s_wait_asynccnt 0x0" ::: "memory");
#endif
#endif
}

enum { EPI_F16_N = 0, EPI_F16_T = 1, EPI_F32_SCALE = 2, EPI_F32_RESID = 3 };

// out = A[M,K] x Bt[N,K]^T (+bias[m]).  Both operands row-major with K
// contiguous.  Block tile 128x128, K-step 32, 8 waves, 2x4 WMMA accs/wave.
template <int MODE>
__global__ __launch_bounds__(256)
void wmma_gemm_k(const h16_t* __restrict__ A, const h16_t* __restrict__ Bt,
                 void* __restrict__ outp,
                 const float* __restrict__ bias,
                 const float* __restrict__ resid,
                 int lda, int ldbT, int ldo, int K, float scale)
{
    __shared__ alignas(16) h16_t shA[2][128 * LDSTR];
    __shared__ alignas(16) h16_t shB[2][128 * LDSTR];

    const int tid = threadIdx.x;
    const int m0 = blockIdx.y * 128;
    const int n0 = blockIdx.x * 128;

    // Both tiles: 128 rows x 32 halves; 512 chunks of 16B; 2 per thread each.
    auto issue = [&](int buf, int k0) {
#pragma unroll
        for (int i = 0; i < 2; ++i) {
            const int c = tid + i * 256;
            const int row = c >> 2;
            const int kc  = (c & 3) * 8;
            async_cp16(A  + (size_t)(m0 + row) * lda  + k0 + kc,
                       &shA[buf][row * LDSTR + kc]);
            async_cp16(Bt + (size_t)(n0 + row) * ldbT + k0 + kc,
                       &shB[buf][row * LDSTR + kc]);
        }
    };

    const int lane = tid & 31;
    const int wid  = tid >> 5;
    const int wm = wid & 3;          // 32-row slab
    const int wn = wid >> 2;         // 64-col slab
    const int lo = lane & 15;
    const int hi = lane >> 4;

    v8f acc[2][4] = {};

    issue(0, 0);

    const int nk = K >> 5;
    for (int ks = 0; ks < nk; ++ks) {
        const int cur = ks & 1;
        const bool more = (ks + 1 < nk);
        if (more) issue(cur ^ 1, (ks + 1) << 5);
        if (more) wait_async4(); else wait_async0();  // tile `cur` landed in LDS
        __syncthreads();                              // ... for every wave

        FragU a[2], b[4];
#pragma unroll
        for (int i = 0; i < 2; ++i) {
            // ISA A 16x32 f16 layout: lane lo -> K {0..7,16..23}; lane hi -> +8
            const h16_t* p = &shA[cur][(wm * 32 + i * 16 + lo) * LDSTR + 8 * hi];
            a[i].u[0] = *(const uint4*)(p);
            a[i].u[1] = *(const uint4*)(p + 16);
        }
#pragma unroll
        for (int j = 0; j < 4; ++j) {
            // ISA B 32x16 f16 layout: lane lo -> K {0..7,16..23}; lane hi -> +8
            const h16_t* p = &shB[cur][(wn * 64 + j * 16 + lo) * LDSTR + 8 * hi];
            b[j].u[0] = *(const uint4*)(p);
            b[j].u[1] = *(const uint4*)(p + 16);
        }
#pragma unroll
        for (int i = 0; i < 2; ++i)
#pragma unroll
            for (int j = 0; j < 4; ++j)
                acc[i][j] = __builtin_amdgcn_wmma_f32_16x16x32_f16(
                    false, a[i].v, false, b[j].v, (short)0, acc[i][j], false, false);

        __syncthreads();   // all waves done reading `cur` before it is refilled
    }

    // Epilogue: D layout = lane -> N (lo), vgpr v -> M = v + 8*hi.
#pragma unroll
    for (int i = 0; i < 2; ++i) {
        const int mbase = m0 + wm * 32 + i * 16 + 8 * hi;
        float bv[8];
#pragma unroll
        for (int v = 0; v < 8; ++v) bv[v] = bias ? bias[mbase + v] : 0.f;
#pragma unroll
        for (int j = 0; j < 4; ++j) {
            const int ng = n0 + wn * 64 + j * 16 + lo;
            if constexpr (MODE == EPI_F16_N) {
                h16_t* out = (h16_t*)outp;
#pragma unroll
                for (int v = 0; v < 8; ++v)
                    out[(size_t)(mbase + v) * ldo + ng] = (h16_t)(acc[i][j][v] + bv[v]);
            } else if constexpr (MODE == EPI_F16_T) {
                h16_t* out = (h16_t*)outp;   // out[N, M]: 8 consecutive m -> b128
                h16_t pack[8];
#pragma unroll
                for (int v = 0; v < 8; ++v) pack[v] = (h16_t)(acc[i][j][v] + bv[v]);
                *(uint4*)(out + (size_t)ng * ldo + mbase) = *(const uint4*)pack;
            } else if constexpr (MODE == EPI_F32_SCALE) {
                float* out = (float*)outp;
#pragma unroll
                for (int v = 0; v < 8; ++v)
                    out[(size_t)(mbase + v) * ldo + ng] = acc[i][j][v] * scale;
            } else { // EPI_F32_RESID
                float* out = (float*)outp;
#pragma unroll
                for (int v = 0; v < 8; ++v)
                    out[(size_t)(mbase + v) * ldo + ng] =
                        resid[(size_t)(mbase + v) * ldo + ng] + acc[i][j][v] + bv[v];
            }
        }
    }
}

// GroupNorm stats per (b, g): mean and rsqrt(var+eps) over 16ch x 4096.
__global__ __launch_bounds__(256)
void gn_stats_k(const float* __restrict__ x, float* __restrict__ mean_out,
                float* __restrict__ rinv_out)
{
    __shared__ float s1[256], s2[256];
    const int tid = threadIdx.x;
    const int b = blockIdx.x >> 5;
    const int g = blockIdx.x & 31;
    const size_t base = ((size_t)b * 512 + (size_t)g * 16) * 4096;
    const int CNT = 16 * 4096;

    float sum = 0.f, sq = 0.f;
    for (int i = tid; i < CNT; i += 256) {
        float v = x[base + i];
        sum += v; sq += v * v;
    }
    s1[tid] = sum; s2[tid] = sq;
    __syncthreads();
    for (int s = 128; s > 0; s >>= 1) {
        if (tid < s) { s1[tid] += s1[tid + s]; s2[tid] += s2[tid + s]; }
        __syncthreads();
    }
    if (tid == 0) {
        const float mean = s1[0] * (1.f / CNT);
        const float var  = s2[0] * (1.f / CNT) - mean * mean;
        mean_out[blockIdx.x] = mean;
        rinv_out[blockIdx.x] = rsqrtf(var + 1e-6f);
    }
}

// Normalize + transpose: x[b,c,n] (f32) -> hT[b,n,c] (f16) via 64x64 LDS tile.
// grid: (N/64, C/64, B)
__global__ __launch_bounds__(256)
void gn_tr_k(const float* __restrict__ x, const float* __restrict__ gw,
             const float* __restrict__ gb, const float* __restrict__ meanv,
             const float* __restrict__ rinvv, h16_t* __restrict__ hT)
{
    __shared__ h16_t t[64][65];
    const int tid = threadIdx.x;
    const int b  = blockIdx.z;
    const int c0 = blockIdx.y * 64;
    const int n0 = blockIdx.x * 64;
    const size_t xbase = ((size_t)b * 512 + c0) * 4096 + n0;

#pragma unroll
    for (int j = 0; j < 16; ++j) {
        const int idx = tid + j * 256;
        const int row = idx >> 6;           // c_local
        const int col = idx & 63;           // n_local
        const int c = c0 + row;
        const int sidx = b * 32 + (c >> 4);
        const float v = x[xbase + (size_t)row * 4096 + col];
        t[row][col] = (h16_t)((v - meanv[sidx]) * rinvv[sidx] * gw[c] + gb[c]);
    }
    __syncthreads();
#pragma unroll
    for (int j = 0; j < 16; ++j) {
        const int idx = tid + j * 256;
        const int nrow = idx >> 6;          // n_local
        const int ccol = idx & 63;          // c_local
        hT[((size_t)b * 4096 + n0 + nrow) * 512 + c0 + ccol] = t[ccol][nrow];
    }
}

// Row softmax: S[qi, ki] f32 -> P[qi, ki] f16, one block per row, values in
// registers (single read + single write of S).
__global__ __launch_bounds__(256)
void softmax_row_k(const float* __restrict__ S, h16_t* __restrict__ P)
{
    __shared__ float red[256];
    const int N = 4096;
    const int tid = threadIdx.x;
    const float* r = S + (size_t)blockIdx.x * N;
    h16_t* o = P + (size_t)blockIdx.x * N;

    float vals[16];
#pragma unroll
    for (int i = 0; i < 16; ++i) vals[i] = r[tid + i * 256];

    float m = -3.4e38f;
#pragma unroll
    for (int i = 0; i < 16; ++i) m = fmaxf(m, vals[i]);
    red[tid] = m;
    __syncthreads();
    for (int s = 128; s > 0; s >>= 1) {
        if (tid < s) red[tid] = fmaxf(red[tid], red[tid + s]);
        __syncthreads();
    }
    m = red[0];
    __syncthreads();

    float sum = 0.f;
#pragma unroll
    for (int i = 0; i < 16; ++i) { vals[i] = __expf(vals[i] - m); sum += vals[i]; }
    red[tid] = sum;
    __syncthreads();
    for (int s = 128; s > 0; s >>= 1) {
        if (tid < s) red[tid] += red[tid + s];
        __syncthreads();
    }
    const float inv = 1.f / red[0];
#pragma unroll
    for (int i = 0; i < 16; ++i) o[tid + i * 256] = (h16_t)(vals[i] * inv);
}

// Convert the four 512x512 f32 weight matrices to f16.
__global__ __launch_bounds__(256)
void cvt_w_k(const float* __restrict__ wq, const float* __restrict__ wk,
             const float* __restrict__ wv, const float* __restrict__ wo,
             h16_t* __restrict__ out)
{
    const int S = 512 * 512;
    const int i = blockIdx.x * 256 + threadIdx.x;
    const float* src = (i < S) ? wq : (i < 2 * S) ? wk : (i < 3 * S) ? wv : wo;
    out[i] = (h16_t)src[i & (S - 1)];
}

extern "C" void kernel_launch(void* const* d_in, const int* in_sizes, int n_in,
                              void* d_out, int out_size, void* d_ws, size_t ws_size,
                              hipStream_t stream)
{
    const float* x  = (const float*)d_in[0];
    const float* gw = (const float*)d_in[1];
    const float* gb = (const float*)d_in[2];
    const float* wq = (const float*)d_in[3];
    const float* bq = (const float*)d_in[4];
    const float* wk = (const float*)d_in[5];
    const float* bk = (const float*)d_in[6];
    const float* wv = (const float*)d_in[7];
    const float* bv = (const float*)d_in[8];
    const float* wo = (const float*)d_in[9];
    const float* bo = (const float*)d_in[10];
    float* out = (float*)d_out;

    const int Bn = 4, C = 512, N = 4096;
    const size_t CN = (size_t)C * N;
    const size_t CC = (size_t)C * C;

    // Scratch (~178 MB): stats | w16 | hT | qT | kT | v | h2T | S f32 | P f16
    float* stats = (float*)d_ws;              // 128 mean + 128 rinv
    h16_t* w16  = (h16_t*)(stats + 256);
    h16_t* wq16 = w16;
    h16_t* wk16 = w16 + CC;
    h16_t* wv16 = w16 + 2 * CC;
    h16_t* wo16 = w16 + 3 * CC;
    h16_t* hT   = w16 + 4 * CC;               // [b][n][c]
    h16_t* qT   = hT  + Bn * CN;              // [b][n][c]
    h16_t* kT   = qT  + Bn * CN;              // [b][n][c]
    h16_t* v16b = kT  + Bn * CN;              // [b][c][n]
    h16_t* h2T  = v16b + Bn * CN;             // [b][n][c]
    float* Sb   = (float*)(h2T + Bn * CN);    // [qi][ki] f32, reused per batch
    h16_t* Pb   = (h16_t*)(Sb + (size_t)N * N);

    cvt_w_k<<<(unsigned)((4 * CC) / 256), 256, 0, stream>>>(wq, wk, wv, wo, w16);
    gn_stats_k<<<Bn * 32, 256, 0, stream>>>(x, stats, stats + 128);
    gn_tr_k<<<dim3(N / 64, C / 64, Bn), 256, 0, stream>>>(
        x, gw, gb, stats, stats + 128, hT);

    const float scl = 22.62741699796952f;  // sqrt(512); reference MULTIPLIES by it
    dim3 blk(256);
    dim3 gProj(N / 128, C / 128);      // (32, 4)  M=512
    dim3 gScore(N / 128, N / 128);     // (32, 32) M=4096

    for (int b = 0; b < Bn; ++b) {
        const h16_t* hb = hT + (size_t)b * CN;
        // qT = (Wq h)^T, kT = (Wk h)^T, v = Wv h (natural)
        wmma_gemm_k<EPI_F16_T><<<gProj, blk, 0, stream>>>(
            wq16, hb, qT + (size_t)b * CN, bq, nullptr, C, C, C, C, 1.f);
        wmma_gemm_k<EPI_F16_T><<<gProj, blk, 0, stream>>>(
            wk16, hb, kT + (size_t)b * CN, bk, nullptr, C, C, C, C, 1.f);
        wmma_gemm_k<EPI_F16_N><<<gProj, blk, 0, stream>>>(
            wv16, hb, v16b + (size_t)b * CN, bv, nullptr, C, C, N, C, 1.f);
        // S[qi,ki] = sqrt(C) * qT[qi,:] . kT[ki,:]
        wmma_gemm_k<EPI_F32_SCALE><<<gScore, blk, 0, stream>>>(
            qT + (size_t)b * CN, kT + (size_t)b * CN, Sb, nullptr, nullptr,
            C, C, N, C, scl);
        // row softmax -> P[qi,ki] f16 (the [N,K] operand of V.P)
        softmax_row_k<<<N, blk, 0, stream>>>(Sb, Pb);
        // h2T[qi,c] = sum_ki v[c,ki] P[qi,ki]
        wmma_gemm_k<EPI_F16_T><<<gProj, blk, 0, stream>>>(
            v16b + (size_t)b * CN, Pb, h2T + (size_t)b * CN, nullptr, nullptr,
            N, N, C, N, 1.f);
        // out = x + Wo h2 + bo  (f32)
        wmma_gemm_k<EPI_F32_RESID><<<gProj, blk, 0, stream>>>(
            wo16, h2T + (size_t)b * CN, out + (size_t)b * CN, bo,
            x + (size_t)b * CN, C, C, N, C, 1.f);
    }
}
